// TransformerBlock_12695923327356
// MI455X (gfx1250) — matrix operand order repaired and checked by reading, not compiled, not run on hardware
//
#include <hip/hip_runtime.h>

// ---------------------------------------------------------------------------
// Transformer block for gfx1250 (MI455X, wave32, WMMA).
// All GEMMs: v_wmma_f32_16x16x32_f16, LDS-tiled + double-buffered with
// GLOBAL_LOAD_ASYNC_TO_LDS (ASYNCcnt) when the toolchain exposes it.
// Attention is flash-style (online softmax), scores never touch memory.
// ---------------------------------------------------------------------------

typedef __attribute__((ext_vector_type(16))) _Float16 v16h;
typedef __attribute__((ext_vector_type(8)))  _Float16 v8h;
typedef __attribute__((ext_vector_type(8)))  float    v8f;

#define NEG_INF (-__builtin_inff())

// Problem constants (match reference)
static constexpr int  CB  = 4;        // batch
static constexpr int  CS  = 2048;     // sequence
static constexpr int  CD  = 1024;     // embed
static constexpr int  CH  = 16;       // heads
static constexpr int  CHS = 64;       // head size
static constexpr int  CNT = CB * CS;  // 8192 tokens
static constexpr int  CFF = 4 * CD;   // 4096

// CDNA5 async global->LDS path (guarded; falls back to reg staging).
// Builtin signature (from clang diagnostic): args are pointers to
// int __attribute__((vector_size(16))) in the global / LDS address spaces.
#if __has_builtin(__builtin_amdgcn_global_load_async_to_lds_b128) && \
    __has_builtin(__builtin_amdgcn_s_wait_asynccnt)
#define ASYNC_LDS 1
typedef int a4i __attribute__((vector_size(4 * sizeof(int))));
typedef __attribute__((address_space(1))) a4i* gas_ptr;
typedef __attribute__((address_space(3))) a4i* las_ptr;
#define GPTR(p) ((gas_ptr)(p))
#define LPTR(p) ((las_ptr)(p))
#else
#define ASYNC_LDS 0
#endif

// ---------------------------------------------------------------------------
// WMMA helpers
// ---------------------------------------------------------------------------
__device__ __forceinline__ v8f wmma_f16(v16h a, v16h b, v8f c) {
  // (neg_a, A, neg_b, B, c_mod, C, reuse_a, reuse_b)
  return __builtin_amdgcn_wmma_f32_16x16x32_f16(false, a, false, b,
                                                (short)0, c, false, false);
}

__device__ __forceinline__ v8f zero8() {
  v8f z;
#pragma unroll
  for (int i = 0; i < 8; ++i) z[i] = 0.0f;
  return z;
}

// A fragment: 16x32 f16. Lane L (r=L&15, hg=L>>4) holds
//   elements 0..7  -> A[r][8*hg .. +7], elements 8..15 -> A[r][16 + 8*hg .. +7]
__device__ __forceinline__ v16h load_a_frag(const _Float16* base, int ld) {
  int lane = threadIdx.x & 31;
  int r = lane & 15, hg = lane >> 4;
  const _Float16* p = base + (size_t)r * ld + 8 * hg;
  v8h lo = *(const v8h*)(p);
  v8h hi = *(const v8h*)(p + 16);
  v16h out;
#pragma unroll
  for (int i = 0; i < 8; ++i) { out[i] = lo[i]; out[i + 8] = hi[i]; }
  return out;
}

// B fragment from Bt (=B^T, [N][K] row-major): lane L (n=L&15, hg=L>>4)
// elements 0..7 -> Bt[n][8*hg .. +7], elements 8..15 -> Bt[n][16 + 8*hg .. +7]
__device__ __forceinline__ v16h load_b_frag(const _Float16* bt, int ld) {
  int lane = threadIdx.x & 31;
  int n = lane & 15, hg = lane >> 4;
  const _Float16* p = bt + (size_t)n * ld + 8 * hg;
  v8h lo = *(const v8h*)(p);
  v8h hi = *(const v8h*)(p + 16);
  v16h out;
#pragma unroll
  for (int i = 0; i < 8; ++i) { out[i] = lo[i]; out[i + 8] = hi[i]; }
  return out;
}

__device__ __forceinline__ float red_max16(float x) {
#pragma unroll
  for (int m = 1; m < 16; m <<= 1) x = fmaxf(x, __shfl_xor(x, m, 32));
  return x;
}
__device__ __forceinline__ float red_sum16(float x) {
#pragma unroll
  for (int m = 1; m < 16; m <<= 1) x += __shfl_xor(x, m, 32);
  return x;
}

// ---------------------------------------------------------------------------
// Prep kernels: precision convert + weight transpose/packing
// ---------------------------------------------------------------------------
__global__ __launch_bounds__(256) void k_cvt_f32_f16(const float* __restrict__ in,
                                                     _Float16* __restrict__ out,
                                                     int n) {
  int i = blockIdx.x * 256 + threadIdx.x;
  if (i < n) out[i] = (_Float16)in[i];
}

// WqkvT[3072][1024]; row n: (which=n>>10, head=(n>>6)&15, e=n&63)
__global__ __launch_bounds__(256) void k_pack_qkvw(const float* __restrict__ Wq,
                                                   const float* __restrict__ Wk,
                                                   const float* __restrict__ Wv,
                                                   _Float16* __restrict__ WT) {
  int i = blockIdx.x * 256 + threadIdx.x;   // over 3072*1024
  int n = i >> 10;
  int k = i & 1023;
  int which = n >> 10;
  int head  = (n >> 6) & 15;
  int e     = n & 63;
  size_t src = (size_t)head * CD * CHS + (size_t)k * CHS + e;
  float v = (which == 0) ? Wq[src] : (which == 1) ? Wk[src] : Wv[src];
  WT[i] = (_Float16)v;
}

// WT[n*K + k] = W[k*N + n]  (W is [K][N] row-major)
__global__ __launch_bounds__(256) void k_transpose_w(const float* __restrict__ W,
                                                     _Float16* __restrict__ WT,
                                                     int K, int N) {
  int i = blockIdx.x * 256 + threadIdx.x;   // over K*N
  int n = i / K;
  int k = i - n * K;
  WT[i] = (_Float16)W[(size_t)k * N + n];
}

// ---------------------------------------------------------------------------
// LDS-tiled WMMA GEMM: C[M,N] = A[M,K] x Bt[N,K]^T  (f16 in, f32 acc).
// Block tile 128x128, 8 waves, each wave 16 rows x 128 cols.
// K staged in 32-wide chunks through double-buffered LDS
// (async global->LDS when available).
// ---------------------------------------------------------------------------
enum { MODE_QKV = 0, MODE_PROJ = 1, MODE_FF1 = 2, MODE_FF2 = 3 };

template <int MODE>
__global__ __launch_bounds__(256) void k_gemm_wmma(
    const _Float16* __restrict__ A, const _Float16* __restrict__ Bt,
    int M, int N, int K,
    const float* __restrict__ bias, const float* __restrict__ res,
    _Float16* __restrict__ outh, float* __restrict__ outf) {
  __shared__ _Float16 sA[2][128 * 32];   // 2 x 8 KB
  __shared__ _Float16 sB[2][128 * 32];   // 2 x 8 KB

  int tid  = threadIdx.x;
  int wave = tid >> 5;
  int lane = tid & 31;
  int row0 = blockIdx.x * 128;
  int col0 = blockIdx.y * 128;

  // Staging: each tile is 128 rows x 32 f16 = 512 chunks of 16B.
  // Thread t moves chunks t and t+256 of each tile.
  // chunk c: row = c>>2, part = c&3 (part*8 f16 within the 32-wide row).
  int r0 = tid >> 2,          p0c = tid & 3;
  int r1 = (tid + 256) >> 2,  p1c = (tid + 256) & 3;

  auto stage = [&](int buf, int ks) {
    const _Float16* gA0 = A  + (size_t)(row0 + r0) * K + ks + p0c * 8;
    const _Float16* gA1 = A  + (size_t)(row0 + r1) * K + ks + p1c * 8;
    const _Float16* gB0 = Bt + (size_t)(col0 + r0) * K + ks + p0c * 8;
    const _Float16* gB1 = Bt + (size_t)(col0 + r1) * K + ks + p1c * 8;
    _Float16* lA0 = &sA[buf][r0 * 32 + p0c * 8];
    _Float16* lA1 = &sA[buf][r1 * 32 + p1c * 8];
    _Float16* lB0 = &sB[buf][r0 * 32 + p0c * 8];
    _Float16* lB1 = &sB[buf][r1 * 32 + p1c * 8];
#if ASYNC_LDS
    __builtin_amdgcn_global_load_async_to_lds_b128(GPTR(gA0), LPTR(lA0), 0, 0);
    __builtin_amdgcn_global_load_async_to_lds_b128(GPTR(gA1), LPTR(lA1), 0, 0);
    __builtin_amdgcn_global_load_async_to_lds_b128(GPTR(gB0), LPTR(lB0), 0, 0);
    __builtin_amdgcn_global_load_async_to_lds_b128(GPTR(gB1), LPTR(lB1), 0, 0);
#else
    *(v8h*)lA0 = *(const v8h*)gA0;
    *(v8h*)lA1 = *(const v8h*)gA1;
    *(v8h*)lB0 = *(const v8h*)gB0;
    *(v8h*)lB1 = *(const v8h*)gB1;
#endif
  };

  v8f acc[8];
#pragma unroll
  for (int j = 0; j < 8; ++j) acc[j] = zero8();

  stage(0, 0);
  int nk = K >> 5;
  for (int kk = 0; kk < nk; ++kk) {
    int buf = kk & 1;
#if ASYNC_LDS
    __builtin_amdgcn_s_wait_asynccnt(0);  // own async loads landed in LDS
#endif
    __syncthreads();  // all waves' loads landed; all done reading buf^1
    if (kk + 1 < nk) stage(buf ^ 1, (kk + 1) * 32);

    v16h a = load_a_frag(&sA[buf][wave * 16 * 32], 32);
#pragma unroll
    for (int j = 0; j < 8; ++j) {
      v16h b = load_b_frag(&sB[buf][j * 16 * 32], 32);
      acc[j] = wmma_f16(a, b, acc[j]);
    }
  }

  // C layout: element v of lane L -> row (v + 8*(L>>4)), col (L&15)
  int rb = (lane >> 4) * 8;
  int cl = lane & 15;
#pragma unroll
  for (int j = 0; j < 8; ++j) {
#pragma unroll
    for (int v = 0; v < 8; ++v) {
      int row = row0 + wave * 16 + rb + v;
      int col = col0 + 16 * j + cl;
      float val = acc[j][v];
      if (MODE == MODE_QKV) {
        // Q,K row-major [B,H,S,HS]; V transposed [B,H,HS,S].
        int which = col >> 10, cc = col & 1023;
        int hh = cc >> 6, e = cc & 63;
        int bb = row >> 11, ss = row & 2047;
        size_t bh = (size_t)bb * CH + hh;
        _Float16 hv = (_Float16)val;
        if (which == 0)
          outh[(bh * CS + ss) * CHS + e] = hv;                        // Q
        else if (which == 1)
          outh[(size_t)CNT * CD + (bh * CS + ss) * CHS + e] = hv;     // K
        else
          outh[(size_t)2 * CNT * CD + (bh * CHS + e) * CS + ss] = hv; // V^T
      } else if (MODE == MODE_PROJ) {
        float y = val + bias[col] + res[(size_t)row * CD + col];
        outf[(size_t)row * CD + col] = y;            // f32 residual carrier
        outh[(size_t)row * CD + col] = (_Float16)y;  // f16 for FFN input
      } else if (MODE == MODE_FF1) {
        float y = val + bias[col];
        y = y > 0.0f ? y : 0.0f;                     // ReLU
        outh[(size_t)row * CFF + col] = (_Float16)y;
      } else {  // MODE_FF2
        float y = val + bias[col] + res[(size_t)row * CD + col];
        outf[(size_t)row * CD + col] = y;            // final f32 output
      }
    }
  }
}

// ---------------------------------------------------------------------------
// Flash attention: one wave per (16-query tile, b, h). Q,K row-major [S,HS],
// V transposed [HS,S]. Online softmax; P converts C-frag -> A-frag layout
// through a 1KB LDS staging buffer (intra-wave LDS is in-order).
// ---------------------------------------------------------------------------
__global__ __launch_bounds__(32) void k_flash_attn(
    const _Float16* __restrict__ Q, const _Float16* __restrict__ K,
    const _Float16* __restrict__ Vt, _Float16* __restrict__ O) {
  __shared__ _Float16 Pl[16 * 32];

  int qi = blockIdx.x;            // 0..127 query tile
  int bh = blockIdx.y;            // 0..63  (b*16 + h)
  int b  = bh >> 4;
  int h  = bh & 15;
  int lane = threadIdx.x;
  int rb = (lane >> 4) * 8;
  int cl = lane & 15;
  int qBase = qi * 16;

  const _Float16* Qp = Q + ((size_t)bh * CS + qBase) * CHS;
  const _Float16* Kp = K + (size_t)bh * CS * CHS;
  const _Float16* Vp = Vt + (size_t)bh * CHS * CS;   // [HS][S]

  v16h qa0 = load_a_frag(Qp, CHS);        // k = 0..31
  v16h qa1 = load_a_frag(Qp + 32, CHS);   // k = 32..63

  v8f o[4];
#pragma unroll
  for (int j = 0; j < 4; ++j) o[j] = zero8();
  float m[8], l[8];
#pragma unroll
  for (int v = 0; v < 8; ++v) { m[v] = NEG_INF; l[v] = 0.0f; }

  const float scale = 32.0f;  // sqrt(D); reference multiplies by sqrt(d)
  int ktEnd = qBase + 15;     // causal horizon

  for (int t0 = 0; t0 <= ktEnd; t0 += 32) {
    v8f s0 = zero8(), s1 = zero8();
    s0 = wmma_f16(qa0, load_b_frag(Kp + (size_t)t0 * CHS, CHS), s0);
    s0 = wmma_f16(qa1, load_b_frag(Kp + (size_t)t0 * CHS + 32, CHS), s0);
    s1 = wmma_f16(qa0, load_b_frag(Kp + (size_t)(t0 + 16) * CHS, CHS), s1);
    s1 = wmma_f16(qa1, load_b_frag(Kp + (size_t)(t0 + 16) * CHS + 32, CHS), s1);

    float p0[8], p1[8], corr[8];
#pragma unroll
    for (int v = 0; v < 8; ++v) {
      int r = rb + v;
      float a0 = s0[v] * scale;
      float a1 = s1[v] * scale;
      if (t0 + cl > qBase + r)      a0 = NEG_INF;   // causal mask
      if (t0 + 16 + cl > qBase + r) a1 = NEG_INF;
      float rmax = red_max16(fmaxf(a0, a1));
      float mn = fmaxf(m[v], rmax);
      float c  = __expf(m[v] - mn);
      float e0 = __expf(a0 - mn);   // exp(-inf) = 0 for masked
      float e1 = __expf(a1 - mn);
      l[v] = l[v] * c + red_sum16(e0 + e1);
      m[v] = mn;
      corr[v] = c;
      p0[v] = e0;
      p1[v] = e1;
    }
#pragma unroll
    for (int j = 0; j < 4; ++j)
#pragma unroll
      for (int v = 0; v < 8; ++v) o[j][v] *= corr[v];

    // C-layout P (16x32, f32) -> LDS -> A-fragment (f16)
#pragma unroll
    for (int v = 0; v < 8; ++v) {
      int r = rb + v;
      Pl[r * 32 + cl]      = (_Float16)p0[v];
      Pl[r * 32 + 16 + cl] = (_Float16)p1[v];
    }
    __builtin_amdgcn_wave_barrier();
    v16h pa = load_a_frag(Pl, 32);
    __builtin_amdgcn_wave_barrier();

#pragma unroll
    for (int j = 0; j < 4; ++j)
      o[j] = wmma_f16(pa, load_b_frag(Vp + (size_t)(16 * j) * CS + t0, CS), o[j]);
  }

#pragma unroll
  for (int j = 0; j < 4; ++j)
#pragma unroll
    for (int v = 0; v < 8; ++v) {
      int token = b * CS + qBase + rb + v;
      int colD  = h * CHS + 16 * j + cl;
      O[(size_t)token * CD + colD] = (_Float16)(o[j][v] / l[v]);
    }
}

// ---------------------------------------------------------------------------
// Host launch
// ---------------------------------------------------------------------------
extern "C" void kernel_launch(void* const* d_in, const int* in_sizes, int n_in,
                              void* d_out, int out_size, void* d_ws, size_t ws_size,
                              hipStream_t stream) {
  (void)in_sizes; (void)n_in; (void)out_size; (void)ws_size;
  const float* x  = (const float*)d_in[0];
  const float* Wq = (const float*)d_in[1];
  const float* Wk = (const float*)d_in[2];
  const float* Wv = (const float*)d_in[3];
  const float* Wp = (const float*)d_in[4];
  const float* bp = (const float*)d_in[5];
  const float* W1 = (const float*)d_in[6];
  const float* b1 = (const float*)d_in[7];
  const float* W2 = (const float*)d_in[8];
  const float* b2 = (const float*)d_in[9];
  float* out = (float*)d_out;

  // Workspace layout (bytes); ~136 MB with aliasing.
  char* W = (char*)d_ws;
  _Float16* Xh     = (_Float16*)(W + 0);          // 16 MB (x as f16)
  _Float16* y1h    = Xh;                           // alias: reused after QKV
  _Float16* WqkvT  = (_Float16*)(W + 16777216);   //  6 MB
  _Float16* WpT    = (_Float16*)(W + 23068672);   //  2 MB
  _Float16* W1T    = (_Float16*)(W + 25165824);   //  8 MB
  _Float16* W2T    = (_Float16*)(W + 33554432);   //  8 MB
  _Float16* Qh     = (_Float16*)(W + 41943040);   // 16 MB [B,H,S,HS]
  _Float16* Hh     = Qh;                           // alias: FFN hidden (64 MB)
  _Float16* Kh     = (_Float16*)(W + 58720256);   // 16 MB [B,H,S,HS]
  _Float16* Vth    = (_Float16*)(W + 75497472);   // 16 MB [B,H,HS,S]
  _Float16* Oh     = (_Float16*)(W + 92274688);   // 16 MB [NT,D]
  float*    y1f    = (float*)   (W + 109051904);  // 32 MB f32 residual

  // 1) x -> f16
  k_cvt_f32_f16<<<(CNT * CD) / 256, 256, 0, stream>>>(x, Xh, CNT * CD);
  // 2) pack/transpose weights into [N,K] f16
  k_pack_qkvw<<<(3 * CD * CD) / 256, 256, 0, stream>>>(Wq, Wk, Wv, WqkvT);
  k_transpose_w<<<(CD * CD) / 256, 256, 0, stream>>>(Wp, WpT, CD, CD);
  k_transpose_w<<<(CD * CFF) / 256, 256, 0, stream>>>(W1, W1T, CD, CFF);
  k_transpose_w<<<(CFF * CD) / 256, 256, 0, stream>>>(W2, W2T, CFF, CD);
  // 3) fused QKV projection: [8192,1024] x [1024,3072]
  k_gemm_wmma<MODE_QKV><<<dim3(CNT / 128, (3 * CD) / 128), 256, 0, stream>>>(
      Xh, WqkvT, CNT, 3 * CD, CD, nullptr, nullptr, Qh, nullptr);
  // 4) causal flash attention
  k_flash_attn<<<dim3(CS / 16, CB * CH), 32, 0, stream>>>(Qh, Kh, Vth, Oh);
  // 5) output projection + bias + x residual -> y1 (f32 + f16)
  k_gemm_wmma<MODE_PROJ><<<dim3(CNT / 128, CD / 128), 256, 0, stream>>>(
      Oh, WpT, CNT, CD, CD, bp, x, y1h, y1f);
  // 6) FFN up + ReLU
  k_gemm_wmma<MODE_FF1><<<dim3(CNT / 128, CFF / 128), 256, 0, stream>>>(
      y1h, W1T, CNT, CFF, CD, b1, nullptr, Hh, nullptr);
  // 7) FFN down + bias + y1 residual -> final f32 output
  k_gemm_wmma<MODE_FF2><<<dim3(CNT / 128, CD / 128), 256, 0, stream>>>(
      Hh, W2T, CNT, CD, CFF, b2, y1f, nullptr, out);
}
